// PreFact_19997367730724
// MI455X (gfx1250) — matrix operand order repaired and checked by reading, not compiled, not run on hardware
//
#include <hip/hip_runtime.h>
#include <hip/hip_bf16.h>
#include <math.h>

#define NUSERS 50000
#define NITEMS 30000
#define NENT   100000
#define NEDGE  500000
#define NEI    1000000
#define DIM    64
#define KDIM   32
#define GAMMA_T 0.2f

typedef __attribute__((ext_vector_type(16))) _Float16 v16h;
typedef __attribute__((ext_vector_type(8)))  float    v8f;

// ---------- helpers ----------
__device__ __forceinline__ float n2n(float x) {
  if (isnan(x)) return 0.0f;
  if (isinf(x)) return x > 0.0f ? 1e4f : 1e-4f;
  return x;
}
__device__ __forceinline__ float sigmoidf(float x) {
  return 1.0f / (1.0f + __expf(-x));
}
__device__ __forceinline__ void atomicMaxF(float* addr, float val) {
  if (val >= 0.0f) atomicMax((int*)addr, __float_as_int(val));
  else             atomicMin((unsigned int*)addr, __float_as_uint(val));
}

// ---------- init ----------
__global__ void init_neginf_kernel(float* p, int n) {
  int i = blockIdx.x * 256 + threadIdx.x;
  if (i < n) p[i] = -INFINITY;
}

// ---------- WMMA GEMM: C[M,64] = A[M,64] @ W[64,64], f16 operands, f32 acc ----------
__global__ __launch_bounds__(128) void gemm64_kernel(const float* __restrict__ A,
                                                     const float* __restrict__ W,
                                                     float* __restrict__ C) {
  const int tid = threadIdx.x;
  const int wave = tid >> 5, lane = tid & 31;
  const int mrow = lane & 15, lhi = lane >> 4;
  const int row0 = blockIdx.x * 16;
  const int ncol = wave * 16 + mrow;          // B-fragment column for this lane
  v8f acc = {};
  const float* arow = A + (size_t)(row0 + mrow) * DIM;
  #pragma unroll
  for (int kb = 0; kb < 64; kb += 32) {
    v16h a, b;
    #pragma unroll
    for (int e = 0; e < 16; ++e) {
      int k = kb + (lhi << 4) + e;
      a[e] = (_Float16)arow[k];
    }
    int kb2 = kb + (lhi << 4);
    #pragma unroll
    for (int e = 0; e < 16; ++e) b[e] = (_Float16)W[(size_t)(kb2 + e) * DIM + ncol];
    acc = __builtin_amdgcn_wmma_f32_16x16x32_f16(false, a, false, b, (short)0, acc, false, false);
  }
  #pragma unroll
  for (int j = 0; j < 8; ++j) {
    int m = j + lhi * 8;
    C[(size_t)(row0 + m) * DIM + ncol] = acc[j];
  }
}

// ---------- attention: scores + segment max ----------
__global__ void attn_score_kernel(const float* __restrict__ Q, const float* __restrict__ K,
                                  const int* __restrict__ users, const int* __restrict__ items,
                                  const float* __restrict__ iew,
                                  float* __restrict__ scores, float* __restrict__ segmax, int n) {
  int e = blockIdx.x * 256 + threadIdx.x;
  if (e >= n) return;
  int u = users[e], it = items[e];
  const float* q = Q + (size_t)u * DIM;
  const float* k = K + (size_t)it * DIM;
  float acc = 0.0f;
  #pragma unroll 8
  for (int d = 0; d < DIM; ++d) acc += q[d] * k[d];
  float sc = acc * 0.125f * iew[e];   // 1/sqrt(64)
  scores[e] = sc;
  atomicMaxF(&segmax[u], sc);
}

__global__ void attn_exp_kernel(const float* __restrict__ segmax, float* __restrict__ scores,
                                float* __restrict__ den, float* __restrict__ cnt,
                                const int* __restrict__ users, int n) {
  int e = blockIdx.x * 256 + threadIdx.x;
  if (e >= n) return;
  int u = users[e];
  float ex = __expf(scores[e] - segmax[u]);
  scores[e] = ex;
  atomicAdd(&den[u], ex);
  atomicAdd(&cnt[u], 1.0f);
}

__global__ void pref_acc_kernel(const float* __restrict__ ex, const float* __restrict__ den,
                                const float* __restrict__ V,
                                const int* __restrict__ users, const int* __restrict__ items,
                                float* __restrict__ upref, long n64) {
  long idx = (long)blockIdx.x * 256 + threadIdx.x;
  if (idx >= n64) return;
  int e = (int)(idx >> 6), d = (int)(idx & 63);
  int u = users[e], it = items[e];
  float alpha = ex[e] / den[u];
  atomicAdd(&upref[(size_t)u * DIM + d], V[(size_t)it * DIM + d] * alpha);
}

// column sum of user_pref / max(cnt,1): one block per column
__global__ __launch_bounds__(256) void colsum_kernel(const float* __restrict__ upref,
                                                     const float* __restrict__ cnt,
                                                     float* __restrict__ s) {
  __shared__ float red[256];
  int d = blockIdx.x;
  float acc = 0.0f;
  for (int u = threadIdx.x; u < NUSERS; u += 256)
    acc += upref[(size_t)u * DIM + d] / fmaxf(cnt[u], 1.0f);
  red[threadIdx.x] = acc;
  __syncthreads();
  for (int off = 128; off > 0; off >>= 1) {
    if (threadIdx.x < off) red[threadIdx.x] += red[threadIdx.x + off];
    __syncthreads();
  }
  if (threadIdx.x == 0) s[d] = red[0];
}

__global__ void w2s_kernel(const float* __restrict__ W2, const float* __restrict__ b2,
                           const float* __restrict__ s, float* __restrict__ w2s,
                           float* __restrict__ b2s) {
  int j = threadIdx.x;  // 64 threads
  float acc = 0.0f;
  for (int d = 0; d < DIM; ++d) acc += W2[(size_t)j * DIM + d] * s[d];
  w2s[j] = acc;
  if (j == 0) {
    float a = 0.0f;
    for (int d = 0; d < DIM; ++d) a += b2[d] * s[d];
    b2s[0] = a;
  }
}

// ---------- fused per-edge MLP + router -> omega (WMMA) ----------
__global__ __launch_bounds__(128) void edge_omega_kernel(
    const int* __restrict__ eh, const int* __restrict__ et, const int* __restrict__ etype,
    const float* __restrict__ ent_real, const float* __restrict__ ent_imag,
    const float* __restrict__ rel_real, const float* __restrict__ rel_imag,
    const float* __restrict__ fp_W1, const float* __restrict__ fp_b1,
    const float* __restrict__ fp_W2, const float* __restrict__ fp_b2,
    const float* __restrict__ rt_W1, const float* __restrict__ rt_b1,
    const float* __restrict__ w2s, const float* __restrict__ b2s,
    float* __restrict__ omega) {
  __shared__ float feat[16][192];
  __shared__ float hid[16][64];
  __shared__ float fact[16][64];
  __shared__ float wrow[16][3];
  __shared__ float oacc[16];
  __shared__ int hs[16], ts[16], rs[16];

  const int tid = threadIdx.x;
  const int base = blockIdx.x * 16;
  if (tid < 16) {
    hs[tid] = eh[base + tid];
    ts[tid] = et[base + tid];
    rs[tid] = etype[base + tid] - 1;
    oacc[tid] = b2s[0];
  }
  __syncthreads();

  // gather feat = [hr hi rr ri tr ti] (16 x 192)
  for (int f = tid; f < 16 * 192; f += 128) {
    int row = f / 192, c = f - row * 192;
    int seg = c >> 5, off = c & 31;
    float v;
    switch (seg) {
      case 0: v = ent_real[(size_t)hs[row] * KDIM + off]; break;
      case 1: v = ent_imag[(size_t)hs[row] * KDIM + off]; break;
      case 2: v = rel_real[(size_t)rs[row] * KDIM + off]; break;
      case 3: v = rel_imag[(size_t)rs[row] * KDIM + off]; break;
      case 4: v = ent_real[(size_t)ts[row] * KDIM + off]; break;
      default: v = ent_imag[(size_t)ts[row] * KDIM + off]; break;
    }
    feat[row][c] = v;
  }
  __syncthreads();

  const int wave = tid >> 5, lane = tid & 31;
  const int mrow = lane & 15, lhi = lane >> 4;
  const int ncol = wave * 16 + mrow;

  // GEMM1: hidden = feat @ fp_W1   (K=192 -> 6 WMMAs)
  v8f c1 = {};
  #pragma unroll
  for (int kb = 0; kb < 192; kb += 32) {
    v16h a, b;
    #pragma unroll
    for (int e = 0; e < 16; ++e) {
      int k = kb + (lhi << 4) + e;
      a[e] = (_Float16)feat[mrow][k];
    }
    int kb2 = kb + (lhi << 4);
    #pragma unroll
    for (int e = 0; e < 16; ++e) b[e] = (_Float16)fp_W1[(size_t)(kb2 + e) * DIM + ncol];
    c1 = __builtin_amdgcn_wmma_f32_16x16x32_f16(false, a, false, b, (short)0, c1, false, false);
  }
  {
    float bn = fp_b1[ncol];
    #pragma unroll
    for (int j = 0; j < 8; ++j) {
      int m = j + lhi * 8;
      hid[m][ncol] = sigmoidf(c1[j] + bn);
    }
  }
  __syncthreads();

  // logits (16x3) + softmax
  if (tid < 48) {
    int row = tid / 3, c = tid - row * 3;
    float acc = fp_b2[c];
    for (int k = 0; k < 64; ++k) acc += hid[row][k] * fp_W2[k * 3 + c];
    wrow[row][c] = acc;
  }
  __syncthreads();
  if (tid < 16) {
    float a0 = wrow[tid][0], a1 = wrow[tid][1], a2 = wrow[tid][2];
    float mx = fmaxf(a0, fmaxf(a1, a2));
    float x0 = __expf(a0 - mx), x1 = __expf(a1 - mx), x2 = __expf(a2 - mx);
    float dn = x0 + x1 + x2;
    wrow[tid][0] = x0 / dn; wrow[tid][1] = x1 / dn; wrow[tid][2] = x2 / dn;
  }
  __syncthreads();

  // fact = [w0*hr+w1*rr+w2*tr , w0*hi+w1*ri+w2*ti]
  for (int f = tid; f < 16 * 64; f += 128) {
    int row = f >> 6, d = f & 63;
    float w0 = wrow[row][0], w1 = wrow[row][1], w2 = wrow[row][2];
    float v;
    if (d < 32) v = w0 * feat[row][d] + w1 * feat[row][64 + d] + w2 * feat[row][128 + d];
    else {
      int dd = d - 32;
      v = w0 * feat[row][32 + dd] + w1 * feat[row][96 + dd] + w2 * feat[row][160 + dd];
    }
    fact[row][d] = v;
  }
  __syncthreads();

  // GEMM2: h2 = fact @ rt_W1  (K=64 -> 2 WMMAs)
  v8f c2 = {};
  #pragma unroll
  for (int kb = 0; kb < 64; kb += 32) {
    v16h a, b;
    #pragma unroll
    for (int e = 0; e < 16; ++e) {
      int k = kb + (lhi << 4) + e;
      a[e] = (_Float16)fact[mrow][k];
    }
    int kb2 = kb + (lhi << 4);
    #pragma unroll
    for (int e = 0; e < 16; ++e) b[e] = (_Float16)rt_W1[(size_t)(kb2 + e) * DIM + ncol];
    c2 = __builtin_amdgcn_wmma_f32_16x16x32_f16(false, a, false, b, (short)0, c2, false, false);
  }
  {
    float bn = rt_b1[ncol], wn = w2s[ncol];
    #pragma unroll
    for (int j = 0; j < 8; ++j) {
      int m = j + lhi * 8;
      atomicAdd(&oacc[m], sigmoidf(c2[j] + bn) * wn);
    }
  }
  __syncthreads();
  if (tid < 16) omega[base + tid] = oacc[tid] * 0.125f;  // / sqrt(HID=64)
}

// ---------- graph conv prep ----------
__global__ void deg_kernel(const int* __restrict__ eh, const float* __restrict__ omega,
                           float* __restrict__ cnt_h, float* __restrict__ seg_o, int n) {
  int e = blockIdx.x * 256 + threadIdx.x;
  if (e >= n) return;
  int h = eh[e];
  atomicAdd(&cnt_h[h], 1.0f);
  atomicAdd(&seg_o[h], omega[e]);
}

__global__ void alpha_kernel(const int* __restrict__ eh, const float* __restrict__ omega,
                             const float* __restrict__ seg_o, float* __restrict__ alpha_g,
                             float* __restrict__ eta0, float* __restrict__ seg_e, int n) {
  int e = blockIdx.x * 256 + threadIdx.x;
  if (e >= n) return;
  int h = eh[e];
  float a = omega[e] / (seg_o[h] + 1e-8f);
  alpha_g[e] = a;
  float et0 = (a > GAMMA_T) ? a : 0.0f;
  eta0[e] = et0;
  atomicAdd(&seg_e[h], et0);
}

// ---------- hops ----------
__global__ void hop_ent_kernel(const int* __restrict__ eh, const int* __restrict__ et,
                               const int* __restrict__ etype,
                               const float* __restrict__ alpha_g, const float* __restrict__ eta0,
                               const float* __restrict__ seg_e,
                               const float* __restrict__ ecur, const float* __restrict__ rel_emb,
                               float* __restrict__ e_agg, long n64, int use_eta) {
  long idx = (long)blockIdx.x * 256 + threadIdx.x;
  if (idx >= n64) return;
  int e = (int)(idx >> 6), d = (int)(idx & 63);
  int h = eh[e], t = et[e], r = etype[e] - 1;
  float rho = use_eta ? (eta0[e] / (seg_e[h] + 1e-8f)) : alpha_g[e];
  float val = rho * ecur[(size_t)t * DIM + d] * rel_emb[(size_t)r * DIM + d];
  atomicAdd(&e_agg[(size_t)h * DIM + d], val);
}

__global__ void hop_usr_kernel(const int* __restrict__ users, const int* __restrict__ items,
                               const float* __restrict__ iew, const float* __restrict__ ecur,
                               float* __restrict__ u_agg, long n64) {
  long idx = (long)blockIdx.x * 256 + threadIdx.x;
  if (idx >= n64) return;
  int e = (int)(idx >> 6), d = (int)(idx & 63);
  int u = users[e], it = items[e];
  atomicAdd(&u_agg[(size_t)u * DIM + d], iew[e] * ecur[(size_t)it * DIM + d]);
}

__global__ __launch_bounds__(64) void norm_ent_kernel(const float* __restrict__ agg,
                                                      const float* __restrict__ cnt_h,
                                                      float* __restrict__ enew,
                                                      float* __restrict__ out_acc) {
  __shared__ float red[64];
  int row = blockIdx.x, d = threadIdx.x;
  float x = agg[(size_t)row * DIM + d] / fmaxf(cnt_h[row], 1.0f);
  red[d] = x * x;
  __syncthreads();
  for (int off = 32; off > 0; off >>= 1) {
    if (d < off) red[d] += red[d + off];
    __syncthreads();
  }
  float y = n2n(x / fmaxf(sqrtf(red[0]), 1e-8f));
  enew[(size_t)row * DIM + d] = y;
  out_acc[(size_t)row * DIM + d] += y;
}

__global__ __launch_bounds__(64) void norm_usr_kernel(const float* __restrict__ agg,
                                                      float* __restrict__ out_acc) {
  __shared__ float red[64];
  int row = blockIdx.x, d = threadIdx.x;
  float x = agg[(size_t)row * DIM + d];
  red[d] = x * x;
  __syncthreads();
  for (int off = 32; off > 0; off >>= 1) {
    if (d < off) red[d] += red[d + off];
    __syncthreads();
  }
  float y = n2n(x / fmaxf(sqrtf(red[0]), 1e-8f));
  out_acc[(size_t)row * DIM + d] += y;
}

// ---------- host ----------
extern "C" void kernel_launch(void* const* d_in, const int* in_sizes, int n_in,
                              void* d_out, int out_size, void* d_ws, size_t ws_size,
                              hipStream_t stream) {
  (void)in_sizes; (void)n_in; (void)out_size; (void)ws_size;
  const float* user_embed = (const float*)d_in[0];
  const float* item_embed = (const float*)d_in[1];
  const float* Wq         = (const float*)d_in[2];
  const float* Wk         = (const float*)d_in[3];
  const float* Wv         = (const float*)d_in[4];
  const float* ent_real   = (const float*)d_in[5];
  const float* ent_imag   = (const float*)d_in[6];
  const float* rel_real   = (const float*)d_in[7];
  const float* rel_imag   = (const float*)d_in[8];
  const float* fp_W1      = (const float*)d_in[9];
  const float* fp_b1      = (const float*)d_in[10];
  const float* fp_W2      = (const float*)d_in[11];
  const float* fp_b2      = (const float*)d_in[12];
  const float* rt_W1      = (const float*)d_in[13];
  const float* rt_b1      = (const float*)d_in[14];
  const float* rt_W2      = (const float*)d_in[15];
  const float* rt_b2      = (const float*)d_in[16];
  const float* rel_emb    = (const float*)d_in[17];
  const float* user_emb0  = (const float*)d_in[18];
  const float* entity_emb = (const float*)d_in[19];
  const float* iew        = (const float*)d_in[20];
  const int*   edge_index = (const int*)d_in[21];
  const int*   edge_type  = (const int*)d_in[22];
  const int*   inter_edge = (const int*)d_in[23];

  const int* eh = edge_index;
  const int* et = edge_index + NEDGE;
  const int* users = inter_edge;
  const int* items = inter_edge + NEI;

  // carve workspace
  char* wsb = (char*)d_ws;
  auto carve = [&](size_t nfloat) -> float* {
    float* p = (float*)wsb;
    wsb += ((nfloat * sizeof(float) + 255) / 256) * 256;
    return p;
  };
  float* Q       = carve((size_t)NUSERS * DIM);
  float* Kmat    = carve((size_t)NITEMS * DIM);
  float* Vmat    = carve((size_t)NITEMS * DIM);
  float* scores  = carve(NEI);
  float* segmax  = carve(NUSERS);
  float* den     = carve(NUSERS);
  float* cnt_u   = carve(NUSERS);
  float* upref   = carve((size_t)NUSERS * DIM);
  float* svec    = carve(64);
  float* w2s     = carve(64);
  float* b2s     = carve(8);
  float* omega   = carve(NEDGE);
  float* alpha_g = carve(NEDGE);
  float* eta0    = carve(NEDGE);
  float* cnt_h   = carve(NENT);
  float* seg_o   = carve(NENT);
  float* seg_e   = carve(NENT);
  float* e_agg   = carve((size_t)NENT * DIM);
  float* u_agg   = carve((size_t)NUSERS * DIM);
  float* eA      = carve((size_t)NENT * DIM);
  float* eB      = carve((size_t)NENT * DIM);

  float* out = (float*)d_out;

  // init output residuals and accumulators
  hipMemcpyAsync(out, entity_emb, (size_t)NENT * DIM * sizeof(float),
                 hipMemcpyDeviceToDevice, stream);
  hipMemcpyAsync(out + (size_t)NENT * DIM, user_emb0, (size_t)NUSERS * DIM * sizeof(float),
                 hipMemcpyDeviceToDevice, stream);
  hipMemsetAsync(den,   0, NUSERS * sizeof(float), stream);
  hipMemsetAsync(cnt_u, 0, NUSERS * sizeof(float), stream);
  hipMemsetAsync(upref, 0, (size_t)NUSERS * DIM * sizeof(float), stream);
  hipMemsetAsync(cnt_h, 0, NENT * sizeof(float), stream);
  hipMemsetAsync(seg_o, 0, NENT * sizeof(float), stream);
  hipMemsetAsync(seg_e, 0, NENT * sizeof(float), stream);
  init_neginf_kernel<<<(NUSERS + 255) / 256, 256, 0, stream>>>(segmax, NUSERS);

  // QKV projections (WMMA)
  gemm64_kernel<<<NUSERS / 16, 128, 0, stream>>>(user_embed, Wq, Q);
  gemm64_kernel<<<NITEMS / 16, 128, 0, stream>>>(item_embed, Wk, Kmat);
  gemm64_kernel<<<NITEMS / 16, 128, 0, stream>>>(item_embed, Wv, Vmat);

  // attention preference miner
  attn_score_kernel<<<(NEI + 255) / 256, 256, 0, stream>>>(Q, Kmat, users, items, iew,
                                                           scores, segmax, NEI);
  attn_exp_kernel<<<(NEI + 255) / 256, 256, 0, stream>>>(segmax, scores, den, cnt_u, users, NEI);
  {
    long n64 = (long)NEI * DIM;
    pref_acc_kernel<<<(int)((n64 + 255) / 256), 256, 0, stream>>>(scores, den, Vmat,
                                                                  users, items, upref, n64);
  }
  colsum_kernel<<<64, 256, 0, stream>>>(upref, cnt_u, svec);
  w2s_kernel<<<1, 64, 0, stream>>>(rt_W2, rt_b2, svec, w2s, b2s);

  // fused fact-semantic + router -> omega (WMMA)
  edge_omega_kernel<<<NEDGE / 16, 128, 0, stream>>>(eh, et, edge_type,
                                                    ent_real, ent_imag, rel_real, rel_imag,
                                                    fp_W1, fp_b1, fp_W2, fp_b2,
                                                    rt_W1, rt_b1, w2s, b2s, omega);

  // graph-conv gate coefficients
  deg_kernel<<<(NEDGE + 255) / 256, 256, 0, stream>>>(eh, omega, cnt_h, seg_o, NEDGE);
  alpha_kernel<<<(NEDGE + 255) / 256, 256, 0, stream>>>(eh, omega, seg_o, alpha_g, eta0,
                                                        seg_e, NEDGE);

  // 3 hops
  const float* ecur = entity_emb;
  float* ebufs[2] = {eA, eB};
  for (int hop = 1; hop <= 3; ++hop) {
    hipMemsetAsync(e_agg, 0, (size_t)NENT * DIM * sizeof(float), stream);
    hipMemsetAsync(u_agg, 0, (size_t)NUSERS * DIM * sizeof(float), stream);
    long ne64 = (long)NEDGE * DIM;
    hop_ent_kernel<<<(int)((ne64 + 255) / 256), 256, 0, stream>>>(
        eh, et, edge_type, alpha_g, eta0, seg_e, ecur, rel_emb, e_agg, ne64,
        (hop >= 3) ? 1 : 0);
    long nu64 = (long)NEI * DIM;
    hop_usr_kernel<<<(int)((nu64 + 255) / 256), 256, 0, stream>>>(users, items, iew, ecur,
                                                                  u_agg, nu64);
    float* enew = ebufs[(hop - 1) & 1];
    norm_ent_kernel<<<NENT, 64, 0, stream>>>(e_agg, cnt_h, enew, out);
    norm_usr_kernel<<<NUSERS, 64, 0, stream>>>(u_agg, out + (size_t)NENT * DIM);
    ecur = enew;
  }
}
